// ResAttnBlock_28054726378329
// MI455X (gfx1250) — matrix operand order repaired and checked by reading, not compiled, not run on hardware
//
#include <hip/hip_runtime.h>
#include <cstdint>
#include <cstddef>

typedef _Float16 h16;
typedef __attribute__((ext_vector_type(16))) _Float16 v16h;
typedef __attribute__((ext_vector_type(8)))  float    v8f;

#define BB   32
#define CC   256
#define NN   1024      // S*S
#define NGRP 32
#define CPG  8         // CC / NGRP

// ---------------------------------------------------------------------------
// WMMA helpers (CDNA5 wave32, v_wmma_f32_16x16x32_f16, f32 accumulate)
// ---------------------------------------------------------------------------
__device__ __forceinline__ v8f wmma16(v16h a, v16h b, v8f c) {
  return __builtin_amdgcn_wmma_f32_16x16x32_f16(
      /*neg_a=*/false, a, /*neg_b=*/false, b,
      /*c_mod=*/(short)0, c, /*reuse_a=*/false, /*reuse_b=*/false);
}

// A-matrix 16x32 f16 fragment from row-major src (ld in elements).
// Lane layout (ISA 7.12.2): lane = row M (mod 16); half-wave selects kbase 0/8;
// elements 0..7 hold K=kb..kb+7, elements 8..15 hold K=16+kb..16+kb+7
// -> two contiguous 16B loads per lane.
__device__ __forceinline__ v16h load_a16x32(const h16* p, int ld, int lane) {
  const int m  = lane & 15;
  const int kb = (lane >> 4) << 3;
  const h16* r = p + (size_t)m * ld + kb;
  v16h a;
  ((uint4*)&a)[0] = *(const uint4*)(r);
  ((uint4*)&a)[1] = *(const uint4*)(r + 16);
  return a;
}

// B-matrix 32x16 f16 fragment where B[k][n] = src[n*ld + k] (transposed read).
// Lane layout: lane = col N (mod 16); half-wave selects kbase 0/8;
// elements 0..7 hold K=kb..kb+7, elements 8..15 hold K=16+kb..16+kb+7 -> two contiguous 16B loads per lane.
__device__ __forceinline__ v16h load_bT32x16(const h16* p, int ld, int lane) {
  const int n  = lane & 15;
  const int kb = (lane >> 4) << 3;
  const h16* r = p + (size_t)n * ld + kb;
  v16h b;
  ((uint4*)&b)[0] = *(const uint4*)(r);
  ((uint4*)&b)[1] = *(const uint4*)(r + 16);
  return b;
}

// ---------------------------------------------------------------------------
// CDNA5 async memory->LDS copy (ASYNCcnt path, bypasses VGPRs).
// Generic shared pointer: addr[31:0] is the LDS byte address (ISA 10.2).
// ---------------------------------------------------------------------------
__device__ __forceinline__ void async_ld_b128(void* lds_ptr, const void* gptr) {
  const uint32_t l = (uint32_t)(uintptr_t)lds_ptr;
  asm volatile("global_load_async_to_lds_b128 %0, %1, off"
               :: "v"(l), "v"(gptr) : "memory");
}
__device__ __forceinline__ void wait_async0() {
  asm volatile("s_wait_asynccnt 0x0" ::: "memory");
}

// ---------------------------------------------------------------------------
// Kernel 0: convert fp32 weights to f16 once per launch
// ---------------------------------------------------------------------------
__global__ void k_cvt(const float* __restrict__ wa, const float* __restrict__ wp,
                      h16* __restrict__ waH, h16* __restrict__ wpH) {
  const int i = blockIdx.x * blockDim.x + threadIdx.x;
  if (i < 3 * CC * CC) waH[i] = (h16)wa[i];
  if (i < CC * CC)     wpH[i] = (h16)wp[i];
}

// ---------------------------------------------------------------------------
// Kernel 1: GroupNorm + NCHW -> (B,N,C) transpose, f16 output
// One block per (batch, group): 8 channels x 1024 spatial = 8192 floats.
// ---------------------------------------------------------------------------
__global__ void k_gn(const float* __restrict__ x, const float* __restrict__ gamma,
                     const float* __restrict__ beta, h16* __restrict__ hN) {
  const int b = blockIdx.x >> 5;
  const int g = blockIdx.x & 31;
  const int t = threadIdx.x;
  const float* xp = x + ((size_t)b * CC + g * CPG) * NN;

  __shared__ float ssum[256], ssq[256];
  float s = 0.f, q = 0.f;
  for (int i = t; i < CPG * NN; i += 256) { float v = xp[i]; s += v; q += v * v; }
  ssum[t] = s; ssq[t] = q; __syncthreads();
  for (int off = 128; off > 0; off >>= 1) {
    if (t < off) { ssum[t] += ssum[t + off]; ssq[t] += ssq[t + off]; }
    __syncthreads();
  }
  const float inv  = 1.f / (float)(CPG * NN);
  const float mean = ssum[0] * inv;
  const float var  = ssq[0] * inv - mean * mean;
  const float rstd = rsqrtf(var + 1e-5f);

  float gm[CPG], bt[CPG];
  #pragma unroll
  for (int ci = 0; ci < CPG; ++ci) {
    gm[ci] = gamma[g * CPG + ci] * rstd;
    bt[ci] = beta[g * CPG + ci] - mean * gm[ci];
  }
  // each thread writes 8 contiguous f16 channels per spatial position (16B store)
  #pragma unroll
  for (int i = 0; i < NN / 256; ++i) {
    const int n = t + 256 * i;
    __align__(16) h16 hv[CPG];
    #pragma unroll
    for (int ci = 0; ci < CPG; ++ci)
      hv[ci] = (h16)(xp[(size_t)ci * NN + n] * gm[ci] + bt[ci]);
    *(uint4*)(hN + ((size_t)b * NN + n) * CC + g * CPG) = *(const uint4*)hv;
  }
}

// ---------------------------------------------------------------------------
// Kernel 2: QKV projection. One 16x16 output tile per wave, K=256 in 8 WMMAs.
// qkv(N,768) = h(N,256) @ w_attn^T + b_attn ; Q pre-scaled by 1/sqrt(C).
// ---------------------------------------------------------------------------
__global__ void k_qkv(const h16* __restrict__ hN, const h16* __restrict__ waH,
                      const float* __restrict__ b_attn,
                      h16* __restrict__ Qb, h16* __restrict__ Kb, h16* __restrict__ Vb) {
  const int lane = threadIdx.x & 31;
  const int tile = blockIdx.x * 8 + (threadIdx.x >> 5);
  const int b  = tile / (64 * 48);
  const int mt = (tile % (64 * 48)) / 48;
  const int nt = tile % 48;

  const h16* A = hN  + ((size_t)b * NN + mt * 16) * CC;
  const h16* W = waH + (size_t)(nt * 16) * CC;   // B[k][n] = w_attn[nt*16+n][k]
  v8f acc = {};
  #pragma unroll
  for (int kk = 0; kk < CC / 32; ++kk)
    acc = wmma16(load_a16x32(A + kk * 32, CC, lane),
                 load_bT32x16(W + kk * 32, CC, lane), acc);

  const int   j     = nt * 16 + (lane & 15);     // output column in [0,768)
  const float bias  = b_attn[j];
  const int   which = j >> 8;                    // 0:Q 1:K 2:V (uniform per tile)
  h16* dst = (which == 0) ? Qb : (which == 1) ? Kb : Vb;
  const float scale = (which == 0) ? 0.0625f : 1.0f;  // 1/sqrt(256) folded into Q
  const int   c     = j & 255;
  #pragma unroll
  for (int r = 0; r < 8; ++r) {                  // C/D layout: row = r + 8*(lane>>4)
    const int n = mt * 16 + r + ((lane >> 4) << 3);
    dst[((size_t)b * NN + n) * CC + c] = (h16)((acc[r] + bias) * scale);
  }
}

// ---------------------------------------------------------------------------
// Kernel 3: flash attention. 128 threads = 4 waves, each wave owns 16 query
// rows; Q (16x256) and O accumulator (16x256 f32) live in VGPRs.
// K tile: async global->LDS (no VGPR round-trip, ASYNCcnt).
// V tile: staged through VGPRs because it is transposed into LDS so the
//         B-fragment loads become contiguous ds_load_b128 (ld=32).
// ---------------------------------------------------------------------------
__launch_bounds__(128)
__global__ void k_attn(const h16* __restrict__ Qb, const h16* __restrict__ Kb,
                       const h16* __restrict__ Vb, h16* __restrict__ Ob) {
  __shared__ __align__(16) h16 Ks [32 * CC];      // (key, c) row-major, 16 KB
  __shared__ __align__(16) h16 VsT[CC * 32];      // (c, key) transposed,  16 KB
  __shared__ __align__(16) h16 Ps [4][16 * 40];   // per-wave P staging, padded ld=40

  const int tid = threadIdx.x, lane = tid & 31, w = tid >> 5;
  const int b  = blockIdx.x >> 4;
  const int q0 = (blockIdx.x & 15) * 64 + w * 16;

  v16h qf[8];
  #pragma unroll
  for (int kk = 0; kk < 8; ++kk)
    qf[kk] = load_a16x32(Qb + ((size_t)b * NN + q0) * CC + kk * 32, CC, lane);

  const v8f vzero = {};
  v8f o[16];
  #pragma unroll
  for (int j = 0; j < 16; ++j) o[j] = vzero;
  float mrun[8], lrun[8];
  #pragma unroll
  for (int r = 0; r < 8; ++r) { mrun[r] = -3.0e38f; lrun[r] = 0.f; }

  h16* Pw = &Ps[w][0];

  for (int kb = 0; kb < NN / 32; ++kb) {
    const size_t base = ((size_t)b * NN + (size_t)kb * 32) * CC;
    // cooperative staging: 32x256 f16 each for K and V
    for (int i = tid; i < 32 * CC / 8; i += 128) {
      const int key = (i * 8) / CC, c = (i * 8) % CC;
      // K: async memory->LDS b128 (ASYNCcnt), row-major
      async_ld_b128(Ks + key * CC + c, Kb + base + (size_t)key * CC + c);
      // V: through VGPRs with transpose into LDS
      __align__(16) h16 tmp[8];
      *(uint4*)tmp = *(const uint4*)(Vb + base + (size_t)key * CC + c);
      #pragma unroll
      for (int e = 0; e < 8; ++e) VsT[(c + e) * 32 + key] = tmp[e];
    }
    if (kb + 1 < NN / 32)   // hint next K block toward L2 (global_prefetch_b8)
      __builtin_prefetch(Kb + base + (size_t)32 * CC + lane * 64, 0, 1);
    wait_async0();          // this wave's async K copies have landed in LDS
    __syncthreads();        // everyone's copies visible to the workgroup

    // S(16x32) = Qscaled . K^T  -> two 16x16 f32 accumulators
    v8f s0 = vzero, s1 = vzero;
    #pragma unroll
    for (int kk = 0; kk < 8; ++kk) {
      s0 = wmma16(qf[kk], load_bT32x16(Ks + kk * 32, CC, lane), s0);
      s1 = wmma16(qf[kk], load_bT32x16(Ks + 16 * CC + kk * 32, CC, lane), s1);
    }

    // online softmax on C/D layout: row = r + 8*(lane>>4) spans 16 lanes of a half
    #pragma unroll
    for (int r = 0; r < 8; ++r) {
      float mx = fmaxf(s0[r], s1[r]);
      #pragma unroll
      for (int off = 8; off > 0; off >>= 1) mx = fmaxf(mx, __shfl_xor(mx, off, 32));
      const float mnew    = fmaxf(mrun[r], mx);
      const float rescale = __expf(mrun[r] - mnew);
      const float p0 = __expf(s0[r] - mnew);
      const float p1 = __expf(s1[r] - mnew);
      float rs = p0 + p1;
      #pragma unroll
      for (int off = 8; off > 0; off >>= 1) rs += __shfl_xor(rs, off, 32);
      lrun[r] = lrun[r] * rescale + rs;
      mrun[r] = mnew;
      #pragma unroll
      for (int j = 0; j < 16; ++j) o[j][r] *= rescale;
      // stage P into A layout via LDS (per-wave region, in-order DS ops)
      const int row = r + ((lane >> 4) << 3), col = lane & 15;
      Pw[row * 40 + col]      = (h16)p0;
      Pw[row * 40 + col + 16] = (h16)p1;
    }

    const v16h pa = load_a16x32(Pw, 40, lane);       // P as 16x32 A fragment
    #pragma unroll
    for (int j = 0; j < 16; ++j)                     // O(16x256) += P . V
      o[j] = wmma16(pa, load_bT32x16(VsT + j * 16 * 32, 32, lane), o[j]);
    __syncthreads();
  }

  #pragma unroll
  for (int r = 0; r < 8; ++r) lrun[r] = 1.f / lrun[r];
  #pragma unroll
  for (int j = 0; j < 16; ++j) {
    #pragma unroll
    for (int r = 0; r < 8; ++r) {
      const int n = q0 + r + ((lane >> 4) << 3);
      Ob[((size_t)b * NN + n) * CC + j * 16 + (lane & 15)] = (h16)(o[j][r] * lrun[r]);
    }
  }
}

// ---------------------------------------------------------------------------
// Kernel 4: output projection + bias + residual, transpose back to (B,C,S,S).
// Per lane the 8 accumulator rows are 8 consecutive n positions of one output
// channel -> residual load + store vectorize to two b128 transactions.
// ---------------------------------------------------------------------------
__global__ void k_proj(const h16* __restrict__ Ob, const h16* __restrict__ wpH,
                       const float* __restrict__ b_proj, const float* __restrict__ x,
                       float* __restrict__ out) {
  const int lane = threadIdx.x & 31;
  const int tile = blockIdx.x * 8 + (threadIdx.x >> 5);
  const int b  = tile / (64 * 16);
  const int mt = (tile % (64 * 16)) / 16;
  const int ct = tile % 16;

  const h16* A = Ob  + ((size_t)b * NN + mt * 16) * CC;
  const h16* W = wpH + (size_t)(ct * 16) * CC;   // B[k][n] = w_proj[ct*16+n][k]
  v8f acc = {};
  #pragma unroll
  for (int kk = 0; kk < CC / 32; ++kk)
    acc = wmma16(load_a16x32(A + kk * 32, CC, lane),
                 load_bT32x16(W + kk * 32, CC, lane), acc);

  const int   c    = ct * 16 + (lane & 15);
  const float bias = b_proj[c];
  const int   n0   = mt * 16 + ((lane >> 4) << 3);      // 8 consecutive n per lane
  const size_t idx = ((size_t)b * CC + c) * NN + n0;    // 32B aligned (n0 % 8 == 0)
  const float4 x0 = *(const float4*)(x + idx);
  const float4 x1 = *(const float4*)(x + idx + 4);
  float4 y0, y1;
  y0.x = x0.x + acc[0] + bias;  y0.y = x0.y + acc[1] + bias;
  y0.z = x0.z + acc[2] + bias;  y0.w = x0.w + acc[3] + bias;
  y1.x = x1.x + acc[4] + bias;  y1.y = x1.y + acc[5] + bias;
  y1.z = x1.z + acc[6] + bias;  y1.w = x1.w + acc[7] + bias;
  *(float4*)(out + idx)     = y0;
  *(float4*)(out + idx + 4) = y1;
}

// ---------------------------------------------------------------------------
// Launch. Workspace layout (<= 65 MiB):
//   [0,16M)  h / O (reused)   [16M,32M) Q   [32M,48M) K   [48M,64M) V
//   [64M, +512K) f16 copies of w_attn, w_proj
// ---------------------------------------------------------------------------
extern "C" void kernel_launch(void* const* d_in, const int* in_sizes, int n_in,
                              void* d_out, int out_size, void* d_ws, size_t ws_size,
                              hipStream_t stream) {
  (void)in_sizes; (void)n_in; (void)out_size; (void)ws_size;
  const float* x      = (const float*)d_in[0];
  /* d_in[1] = t_embd: unused by the reference computation */
  const float* gamma  = (const float*)d_in[2];
  const float* beta   = (const float*)d_in[3];
  const float* w_attn = (const float*)d_in[4];
  const float* b_attn = (const float*)d_in[5];
  const float* w_proj = (const float*)d_in[6];
  const float* b_proj = (const float*)d_in[7];
  float* out = (float*)d_out;

  char* ws = (char*)d_ws;
  h16* hO  = (h16*)ws;                                   // h, later reused as O
  h16* Qb  = (h16*)(ws + ((size_t)16 << 20));
  h16* Kb  = (h16*)(ws + ((size_t)32 << 20));
  h16* Vb  = (h16*)(ws + ((size_t)48 << 20));
  h16* waH = (h16*)(ws + ((size_t)64 << 20));
  h16* wpH = waH + 3 * CC * CC;

  k_cvt <<<(3 * CC * CC + 255) / 256, 256, 0, stream>>>(w_attn, w_proj, waH, wpH);
  k_gn  <<<BB * NGRP,            256, 0, stream>>>(x, gamma, beta, hO);
  k_qkv <<<BB * 64 * 48 / 8,     256, 0, stream>>>(hO, waH, b_attn, Qb, Kb, Vb);
  k_attn<<<BB * 16,              128, 0, stream>>>(Qb, Kb, Vb, hO);
  k_proj<<<BB * 64 * 16 / 8,     256, 0, stream>>>(hO, wpH, b_proj, x, out);
}
